// MultiHeadedAttentionWithAdapterCache_65773129171684
// MI455X (gfx1250) — matrix operand order repaired and checked by reading, not compiled, not run on hardware
//
#include <hip/hip_runtime.h>

// ---------------------------------------------------------------------------
// MHA forward for MI455X (gfx1250, wave32, WMMA).
// Stages: (1) weight fp32->bf16, (2) Q/K/V projection GEMMs (bf16 WMMA, fp32
// accum), (3) causal flash-attention with 64-key steps (exp2-domain online
// softmax, DPP max reduction, ones-column WMMA denominator, per-wave trip
// counts with a single peeled masked diagonal block), (4) output GEMM.
// Roofline: ~137 GFLOP, ~0.3 GB moved -> both compute (bf16 WMMA) and HBM
// (23.3 TB/s) land at tens of us. 64-key steps halve the per-step softmax
// overhead (one DPP tree / one corr rescale per row per 64 keys) while the
// per-key WMMA and exp work is unchanged.
// ---------------------------------------------------------------------------

typedef __bf16 bf16;
typedef __attribute__((ext_vector_type(16))) __bf16 v16bf;
typedef __attribute__((ext_vector_type(8)))  __bf16 v8bf;
typedef __attribute__((ext_vector_type(8)))  float  v8f;

constexpr int B_  = 4;
constexpr int S_  = 2048;
constexpr int D_  = 1024;
constexpr int H_  = 16;
constexpr int DH_ = 64;
constexpr size_t DD_   = (size_t)D_ * D_;            // 1,048,576
constexpr size_t BHSD_ = (size_t)B_ * H_ * S_ * DH_; // 8,388,608 (== B*S*D)

union BF16x16 { v16bf v; v8bf h[2]; bf16 e[16]; };

__device__ inline v8f wmma_bf16(v16bf a, v16bf b, v8f c) {
  // D = A(16x32 bf16) x B(32x16 bf16) + C(16x16 f32)
  return __builtin_amdgcn_wmma_f32_16x16x32_bf16(false, a, false, b,
                                                 (short)0, c, false, false);
}

// Max-reduce across the 16-lane half that holds one C-matrix row, entirely in
// VALU via DPP (no ds_bpermute): xor1/xor2 = quad_perm, xor4 = DPP8,
// xor8 = row_ror:8. All patterns stay inside each 16-lane DPP row.
__device__ inline float rowmax16_dpp(float x) {
  int t;
  t = __builtin_amdgcn_update_dpp(0, __float_as_int(x), 0xB1, 0xF, 0xF, true);
  x = fmaxf(x, __int_as_float(t));                       // lanes ^1
  t = __builtin_amdgcn_update_dpp(0, __float_as_int(x), 0x4E, 0xF, 0xF, true);
  x = fmaxf(x, __int_as_float(t));                       // lanes ^2
  t = __builtin_amdgcn_mov_dpp8(__float_as_int(x), 0x688FAC);
  x = fmaxf(x, __int_as_float(t));                       // lanes ^4
  t = __builtin_amdgcn_update_dpp(0, __float_as_int(x), 0x128, 0xF, 0xF, true);
  x = fmaxf(x, __int_as_float(t));                       // lanes ^8 (row_ror:8)
  return x;
}

// A-fragment (16x32, row-major source). Lane half h holds K = h*16+e
// of row (lane&15). `row` is pre-offset to kk.
__device__ inline v16bf load_a_bf16(const bf16* __restrict__ row, int half) {
  BF16x16 u;
  u.h[0] = *(const v8bf*)(row + half * 16);
  u.h[1] = *(const v8bf*)(row + half * 16 + 8);
  return u.v;
}

__device__ inline v16bf load_a_f32(const float* __restrict__ row, int half) {
  const float4 f0 = *(const float4*)(row + half * 16);
  const float4 f1 = *(const float4*)(row + half * 16 + 4);
  const float4 f2 = *(const float4*)(row + half * 16 + 8);
  const float4 f3 = *(const float4*)(row + half * 16 + 12);
  float t[16] = {f0.x, f0.y, f0.z, f0.w, f1.x, f1.y, f1.z, f1.w,
                 f2.x, f2.y, f2.z, f2.w, f3.x, f3.y, f3.z, f3.w};
  BF16x16 u;
#pragma unroll
  for (int i = 0; i < 16; ++i) u.e[i] = (bf16)t[i];
  return u.v;
}

// ---------------------------------------------------------------------------
__global__ void f32_to_bf16_kernel(const float* __restrict__ s,
                                   bf16* __restrict__ d, int n) {
  int i = blockIdx.x * blockDim.x + threadIdx.x;
  if (i < n) d[i] = (bf16)s[i];
}

// ---------------------------------------------------------------------------
// y[m][n] = sum_k X[m][k] * W[n][k] + bias[n]   (torch Linear: X @ W^T + b)
// MODE 0: q  -> bf16 split-head [B,H,S,DH]
// MODE 1: k  -> bf16 split-head + fp32 split-head (d_out)
// MODE 2: v  -> fp32 split-head (d_out) + bf16 transposed [B,H,DH,S]
// MODE 3: out-> fp32 [B,S,D], input X is bf16 (ctx)
// Block: 256 thr = 8 waves as 4(M) x 2(N) -> 64x32 tile. Grid (128, 32).
template <int MODE>
__global__ void __launch_bounds__(256) gemm_bf16_wmma(
    const float* __restrict__ Xf, const bf16* __restrict__ Xb,
    const bf16* __restrict__ W, const float* __restrict__ bias,
    bf16* __restrict__ obf, float* __restrict__ of32, bf16* __restrict__ obfT)
{
  const int lane = threadIdx.x & 31;
  const int wid  = threadIdx.x >> 5;
  const int wm   = wid >> 1, wn = wid & 1;
  const int half = lane >> 4, l16 = lane & 15;
  const int m0 = blockIdx.x * 64 + wm * 16;
  const int n0 = blockIdx.y * 32 + wn * 16;
  const int rowA = m0 + l16;   // X row this lane contributes to A-frag
  const int rowB = n0 + l16;   // W row (== output column) for B-frag

  const bf16* wrow = W + (size_t)rowB * D_;
  v8f acc = {0.f, 0.f, 0.f, 0.f, 0.f, 0.f, 0.f, 0.f};
#pragma unroll 4
  for (int kk = 0; kk < D_; kk += 32) {
    v16bf a;
    if constexpr (MODE == 3) a = load_a_bf16(Xb + (size_t)rowA * D_ + kk, half);
    else                     a = load_a_f32 (Xf + (size_t)rowA * D_ + kk, half);
    // B[k][n] = W[n][k]: lane reads 16 contiguous bf16 of W row n.
    const v16bf b = *(const v16bf*)(wrow + kk + half * 16);
    acc = wmma_bf16(a, b, acc);
  }

  const float bv = bias[n0 + l16];
  const int n  = n0 + l16;
  const int hh = n >> 6, dd = n & (DH_ - 1);
#pragma unroll
  for (int j = 0; j < 8; ++j) {
    const int   m   = m0 + j + 8 * half;       // C layout: M = j + 8*(lane>>4)
    const float val = acc[j] + bv;
    const int bb = m >> 11, ss = m & (S_ - 1);
    const size_t sh = (((size_t)bb * H_ + hh) * S_ + ss) * DH_ + dd;
    if constexpr (MODE == 0) {
      obf[sh] = (bf16)val;
    } else if constexpr (MODE == 1) {
      obf[sh] = (bf16)val; of32[sh] = val;
    } else if constexpr (MODE == 2) {
      of32[sh] = val;
      obfT[(((size_t)bb * H_ + hh) * DH_ + dd) * S_ + ss] = (bf16)val;
    } else {
      of32[(size_t)m * D_ + n] = val;
    }
  }
}

// ---------------------------------------------------------------------------
// One 64-key flash-attention step for one wave: 8 score WMMAs (4 key groups x
// 2 K-halves), one softmax pass (single DPP tree + single corr rescale per
// row), LDS re-layout of P into two A-fragments, 8 PV WMMAs + 2 ones-WMMAs.
// MASKED=true only for the peeled diagonal block.
template <bool MASKED>
__device__ __forceinline__ void attn_step64(
    int kbase, int qr, int half, int l16,
    const bf16* __restrict__ Kb, const bf16* __restrict__ Vb,
    const v16bf& qa0, const v16bf& qa1, const v16bf& ones_frag,
    v8f cacc[5], float mrow[8], bf16 (* __restrict__ pl)[72])
{
  constexpr float SCL = 0.18033688011112042f;   // 0.125 * log2(e)
  const v8f zero8 = {0.f, 0.f, 0.f, 0.f, 0.f, 0.f, 0.f, 0.f};

  // ---- S = Q K^T for 64 keys: four 16x16 accumulators ---------------------
  v8f s[4] = {zero8, zero8, zero8, zero8};
#pragma unroll
  for (int g = 0; g < 4; ++g) {
    const bf16* kr = Kb + (size_t)(kbase + g * 16 + l16) * DH_;
    const v16bf blo = *(const v16bf*)(kr + half * 16);
    const v16bf bhi = *(const v16bf*)(kr + 32 + half * 16);
    s[g] = wmma_bf16(qa0, blo, s[g]);
    s[g] = wmma_bf16(qa1, bhi, s[g]);
  }

  // ---- scale, (diagonal-only) causal mask, online softmax -----------------
#pragma unroll
  for (int j = 0; j < 8; ++j) {
    float x[4];
#pragma unroll
    for (int g = 0; g < 4; ++g) x[g] = s[g][j] * SCL;
    if constexpr (MASKED) {
      const int qg = qr + j + 8 * half;          // global query row
#pragma unroll
      for (int g = 0; g < 4; ++g)
        if (kbase + g * 16 + l16 > qg) x[g] = -1e30f;
    }
    const float bm =
        rowmax16_dpp(fmaxf(fmaxf(x[0], x[1]), fmaxf(x[2], x[3])));
    const float mn   = fmaxf(mrow[j], bm);
    const float corr = exp2f(mrow[j] - mn);
    mrow[j] = mn;
#pragma unroll
    for (int t = 0; t < 5; ++t) cacc[t][j] *= corr;   // incl. denom tile
    // C-layout -> LDS (row = M, col = local key)
#pragma unroll
    for (int g = 0; g < 4; ++g)
      pl[j + 8 * half][g * 16 + l16] = (bf16)exp2f(x[g] - mn);
  }

  // Wave-private LDS exchange: wave32 DS ops are in-order; wave_barrier only
  // pins the scheduler, no cross-wave sync needed.
  __builtin_amdgcn_wave_barrier();
  BF16x16 p0u, p1u;
  p0u.h[0] = *(const v8bf*)(&pl[l16][half * 16]);
  p0u.h[1] = *(const v8bf*)(&pl[l16][half * 16 + 8]);
  p1u.h[0] = *(const v8bf*)(&pl[l16][32 + half * 16]);
  p1u.h[1] = *(const v8bf*)(&pl[l16][40 + half * 16]);
  const v16bf pa0 = p0u.v;                      // P keys 0..31
  const v16bf pa1 = p1u.v;                      // P keys 32..63
  __builtin_amdgcn_wave_barrier();

  // ---- ctx += P (16x64) @ V (64x64), plus denominator via ones tile -------
#pragma unroll
  for (int t = 0; t < 4; ++t) {
    const bf16* vrow = Vb + (size_t)(t * 16 + l16) * S_ + kbase;
    cacc[t] = wmma_bf16(pa0, *(const v16bf*)(vrow + half * 16), cacc[t]);
    cacc[t] = wmma_bf16(pa1, *(const v16bf*)(vrow + 32 + half * 16), cacc[t]);
  }
  cacc[4] = wmma_bf16(pa0, ones_frag, cacc[4]); // l += rowsum(P[:,0:32])
  cacc[4] = wmma_bf16(pa1, ones_frag, cacc[4]); // l += rowsum(P[:,32:64])
}

// Causal flash attention. Block = 4 fully independent waves (128 thr); wave
// owns 16 q-rows, block owns 64. Each wave runs its own trip count: mask-free
// main loop over full 64-key blocks, then one peeled masked diagonal block.
__global__ void __launch_bounds__(128) attn_fmha(
    const bf16* __restrict__ Q, const bf16* __restrict__ K,
    const bf16* __restrict__ VT, bf16* __restrict__ CTX)
{
  const int lane = threadIdx.x & 31;
  const int wid  = threadIdx.x >> 5;              // 0..3
  const int half = lane >> 4, l16 = lane & 15;
  const int bh = blockIdx.y;
  const int b  = bh >> 4, h = bh & (H_ - 1);
  const int q0 = blockIdx.x * 64;
  const int qr = q0 + wid * 16;

  const bf16* Qb = Q  + (size_t)bh * S_ * DH_;    // [S, DH]
  const bf16* Kb = K  + (size_t)bh * S_ * DH_;    // [S, DH]
  const bf16* Vb = VT + (size_t)bh * DH_ * S_;    // [DH, S] (transposed)

  __shared__ alignas(16) bf16 Pl[4][16][72];      // wave-private slices, padded
  bf16 (* __restrict__ pl)[72] = Pl[wid];

  // Q A-fragments for this wave's 16 rows (K=0..31, K=32..63), kept resident.
  const v16bf qa0 = load_a_bf16(Qb + (size_t)(qr + l16) * DH_, half);
  const v16bf qa1 = load_a_bf16(Qb + (size_t)(qr + l16) * DH_ + 32, half);

  // Constant all-ones B-fragment: every column of ctx tile 4 becomes the row
  // sum of P, i.e. every lane holds its own row's softmax denominator.
  BF16x16 ou;
#pragma unroll
  for (int i = 0; i < 16; ++i) ou.e[i] = (bf16)1.0f;
  const v16bf ones_frag = ou.v;

  const v8f zero8 = {0.f, 0.f, 0.f, 0.f, 0.f, 0.f, 0.f, 0.f};
  v8f cacc[5] = {zero8, zero8, zero8, zero8, zero8}; // ctx 16x64 + denom tile
  float mrow[8];
#pragma unroll
  for (int j = 0; j < 8; ++j) mrow[j] = -1e30f;

  // This wave needs keys [0, qr+16); nfull 64-key blocks are provably below
  // the diagonal, one peeled masked block covers the rest (kbase <= 1984, so
  // the tail never reads past S).
  const int nfull = qr >> 6;
  for (int kb = 0; kb < nfull; ++kb) {
    const int kbase = kb * 64;
    // Prefetch next 64-key block's K rows (gfx1250 global_prefetch path).
    const bf16* nk = Kb + (size_t)(kbase + 64 + l16) * DH_;
    __builtin_prefetch(nk, 0, 3);
    __builtin_prefetch(nk + 16 * DH_, 0, 3);
    __builtin_prefetch(nk + 32 * DH_, 0, 3);
    __builtin_prefetch(nk + 48 * DH_, 0, 3);
    attn_step64<false>(kbase, qr, half, l16, Kb, Vb, qa0, qa1, ones_frag,
                       cacc, mrow, pl);
  }
  attn_step64<true>(nfull * 64, qr, half, l16, Kb, Vb, qa0, qa1, ones_frag,
                    cacc, mrow, pl);

  // ---- normalize, write ctx merged-head [B,S,D] as bf16 -------------------
#pragma unroll
  for (int j = 0; j < 8; ++j) {
    const float inv = 1.0f / cacc[4][j];          // every lane has its row sum
    const int m = qr + j + 8 * half;
    bf16* dst = CTX + ((size_t)b * S_ + m) * D_ + h * DH_;
#pragma unroll
    for (int t = 0; t < 4; ++t) dst[t * 16 + l16] = (bf16)(cacc[t][j] * inv);
  }
}

// ---------------------------------------------------------------------------
extern "C" void kernel_launch(void* const* d_in, const int* in_sizes, int n_in,
                              void* d_out, int out_size, void* d_ws,
                              size_t ws_size, hipStream_t stream) {
  (void)in_sizes; (void)n_in; (void)out_size; (void)ws_size;

  const float* query = (const float*)d_in[0];
  const float* key   = (const float*)d_in[1];
  const float* value = (const float*)d_in[2];
  // d_in[3] is the causal tril mask; causality is applied analytically.
  const float* Wq = (const float*)d_in[4];
  const float* bq = (const float*)d_in[5];
  const float* Wk = (const float*)d_in[6];
  const float* bk = (const float*)d_in[7];
  const float* Wv = (const float*)d_in[8];
  const float* bv = (const float*)d_in[9];
  const float* Wo = (const float*)d_in[10];
  const float* bo = (const float*)d_in[11];

  float* out_p = (float*)d_out;          // [B,S,D]
  float* k_out = out_p + BHSD_;          // [B,H,S,DH]
  float* v_out = k_out + BHSD_;          // [B,H,S,DH]

  bf16* ws   = (bf16*)d_ws;              // ~75.5 MB of bf16 staging
  bf16* Wqb  = ws;
  bf16* Wkb  = Wqb + DD_;
  bf16* Wvb  = Wkb + DD_;
  bf16* Wob  = Wvb + DD_;
  bf16* Qb   = Wob + DD_;                // [B,H,S,DH]
  bf16* Kb   = Qb + BHSD_;               // [B,H,S,DH]
  bf16* VTb  = Kb + BHSD_;               // [B,H,DH,S]
  bf16* CTXb = VTb + BHSD_;              // [B,S,D]

  const int cvtBlocks = (int)(DD_ / 256);
  f32_to_bf16_kernel<<<cvtBlocks, 256, 0, stream>>>(Wq, Wqb, (int)DD_);
  f32_to_bf16_kernel<<<cvtBlocks, 256, 0, stream>>>(Wk, Wkb, (int)DD_);
  f32_to_bf16_kernel<<<cvtBlocks, 256, 0, stream>>>(Wv, Wvb, (int)DD_);
  f32_to_bf16_kernel<<<cvtBlocks, 256, 0, stream>>>(Wo, Wob, (int)DD_);

  const dim3 gg((B_ * S_) / 64, D_ / 32);  // (128, 32)
  gemm_bf16_wmma<0><<<gg, 256, 0, stream>>>(query, nullptr, Wqb, bq, Qb, nullptr, nullptr);
  gemm_bf16_wmma<1><<<gg, 256, 0, stream>>>(key,   nullptr, Wkb, bk, Kb, k_out, nullptr);
  gemm_bf16_wmma<2><<<gg, 256, 0, stream>>>(value, nullptr, Wvb, bv, nullptr, v_out, VTb);

  attn_fmha<<<dim3(S_ / 64, B_ * H_), 128, 0, stream>>>(Qb, Kb, VTb, CTXb);

  gemm_bf16_wmma<3><<<gg, 256, 0, stream>>>(nullptr, CTXb, Wob, bo, nullptr, out_p, nullptr);
}
